// GAEGCN_41197326303335
// MI455X (gfx1250) — matrix operand order repaired and checked by reading, not compiled, not run on hardware
//
#include <hip/hip_runtime.h>
#include <hip/hip_bf16.h>
#include <stdint.h>

#define N_HIDDEN 1024
#define N_CLASS  64

typedef __attribute__((ext_vector_type(16))) __bf16 v16bf;
typedef __attribute__((ext_vector_type(8)))  float  v8f;

union V16U { uint4 u[2]; v16bf v; };

// ---------------- degree / normalization ----------------

__global__ void gcn_deg_init(float* __restrict__ deg, int n) {
    int i = blockIdx.x * blockDim.x + threadIdx.x;
    if (i < n) deg[i] = 1.0f;                    // self-loop contribution
}

__global__ void gcn_deg_edges(const int* __restrict__ ei, float* __restrict__ deg, int nE) {
    int e = blockIdx.x * blockDim.x + threadIdx.x;
    if (e < nE) atomicAdd(&deg[ei[nE + e]], 1.0f);   // dst row of edge_index
}

__global__ void gcn_deg_rsqrt(float* __restrict__ deg, int n) {
    int i = blockIdx.x * blockDim.x + threadIdx.x;
    if (i < n) deg[i] = rsqrtf(deg[i]);          // deg >= 1 always
}

// ---------------- W (fp32 row-major [K][64]) -> bf16 col-major [64][K] ----------------

__global__ void gcn_wcvt(const float* __restrict__ W, __bf16* __restrict__ Wcm) {
    int i = blockIdx.x * blockDim.x + threadIdx.x;   // 0 .. 64*1024-1
    int n = i >> 10;        // output column (class)
    int k = i & 1023;       // input feature
    Wcm[i] = (__bf16)W[k * N_CLASS + n];             // Wcm[n*1024 + k]
}

// ---------------- GEMM: h = x @ W  via v_wmma_f32_16x16x32_bf16 ----------------
// One wave -> 16 rows x 64 cols tile. LDS stages W in K=256 chunks.

#define LDS_STRIDE 280   // 256 + 24 pad: 560B rows -> 16B aligned, conflict-free over 16 lanes

__global__ __launch_bounds__(256)
void gcn_gemm(const float* __restrict__ x, const __bf16* __restrict__ Wcm,
              float* __restrict__ h, int nNodes) {
    __shared__ __bf16 lw[64 * LDS_STRIDE];           // 35840 B

    const int tid  = threadIdx.x;
    const int wid  = tid >> 5;
    const int lane = tid & 31;
    const int g    = lane >> 4;      // half-wave group
    const int m    = lane & 15;      // row within tile / col within N-subtile
    const int tile_m = (blockIdx.x * 8 + wid) * 16;

    int r = tile_m + m;
    const int rl = (r < nNodes) ? r : (nNodes - 1);  // clamp: keep EXEC full for WMMA
    const float* xrow = x + (size_t)rl * N_HIDDEN;

    v8f acc[4];
#pragma unroll
    for (int t = 0; t < 4; ++t) acc[t] = (v8f){0.f,0.f,0.f,0.f,0.f,0.f,0.f,0.f};

    for (int kc = 0; kc < 4; ++kc) {                 // K chunks of 256
        __syncthreads();
        // stage W chunk into LDS: 64 cols x 256 K (bf16) = 2048 uint4; loop issues 4096 (n 0..127)
#pragma unroll
        for (int j = 0; j < 16; ++j) {
            int idx = tid + j * 256;                 // 0..4095
            int n   = idx >> 5;                      // class column 0..127; lw and Wcm hold only 0..63
            int kk8 = (idx & 31) << 3;               // K offset 0..248 step 8
            *reinterpret_cast<uint4*>(&lw[n * LDS_STRIDE + kk8]) =
                *reinterpret_cast<const uint4*>(Wcm + (size_t)n * N_HIDDEN + kc * 256 + kk8);
        }
        __syncthreads();

#pragma unroll
        for (int ks = 0; ks < 8; ++ks) {             // 8 WMMA K-steps of 32
            const int kb = kc * 256 + ks * 32;

            // A operand: ISA 16-bit A layout -> lane holds K = {b*16 + g*8 + 0..7}, b=0,1
            const float4* xa = reinterpret_cast<const float4*>(xrow + kb + g * 8);
            float4 f0 = xa[0];
            float4 f1 = xa[1];
            const float4* xb = reinterpret_cast<const float4*>(xrow + kb + 16 + g * 8);
            float4 f2 = xb[0];
            float4 f3 = xb[1];
            v16bf a;
            a[0]=(__bf16)f0.x;  a[1]=(__bf16)f0.y;  a[2]=(__bf16)f0.z;  a[3]=(__bf16)f0.w;
            a[4]=(__bf16)f1.x;  a[5]=(__bf16)f1.y;  a[6]=(__bf16)f1.z;  a[7]=(__bf16)f1.w;
            a[8]=(__bf16)f2.x;  a[9]=(__bf16)f2.y;  a[10]=(__bf16)f2.z; a[11]=(__bf16)f2.w;
            a[12]=(__bf16)f3.x; a[13]=(__bf16)f3.y; a[14]=(__bf16)f3.z; a[15]=(__bf16)f3.w;

            const int kk = ks * 32 + g * 8;          // B operand: lane holds K = {b*16 + g*8 + 0..7}, b=0,1
#pragma unroll
            for (int nt = 0; nt < 4; ++nt) {
                const int n = nt * 16 + m;
                V16U b;
                b.u[0] = *reinterpret_cast<const uint4*>(&lw[n * LDS_STRIDE + kk]);
                b.u[1] = *reinterpret_cast<const uint4*>(&lw[n * LDS_STRIDE + kk + 16]);
                acc[nt] = __builtin_amdgcn_wmma_f32_16x16x32_bf16(
                    false, a, false, b.v, (short)0, acc[nt], false, false);
            }
        }
    }

    // C/D layout: VGPR vr, lane l -> M = vr + 8*(l/16), N = l%16
#pragma unroll
    for (int vr = 0; vr < 8; ++vr) {
        int row = tile_m + vr + 8 * g;
        if (row < nNodes) {
            float* hp = h + (size_t)row * N_CLASS + m;
            hp[0]  = acc[0][vr];
            hp[16] = acc[1][vr];
            hp[32] = acc[2][vr];
            hp[48] = acc[3][vr];
        }
    }
}

// ---------------- out = h * dis^2 (self loop) + bias ----------------

__global__ void gcn_out_init(const float* __restrict__ h, const float* __restrict__ dis,
                             const float* __restrict__ bias, float* __restrict__ out, int total) {
    int i = blockIdx.x * blockDim.x + threadIdx.x;
    if (i >= total) return;
    int r = i >> 6;
    int c = i & 63;
    float d = dis[r];
    out[i] = h[i] * d * d + bias[c];
}

// ---------------- edge scatter: out[dst] += h[src] * dis[src]*dis[dst] ----------------
// one wave per edge, each lane handles channels {lane, lane+32} (coalesced 128B)

__global__ __launch_bounds__(256)
void gcn_scatter(const int* __restrict__ ei, const float* __restrict__ h,
                 const float* __restrict__ dis, float* __restrict__ out, int nE) {
    int e = (int)((blockIdx.x * blockDim.x + threadIdx.x) >> 5);
    if (e >= nE) return;
    int lane = threadIdx.x & 31;
    int s = ei[e];
    int d = ei[nE + e];
    float norm = dis[s] * dis[d];
    const float* hs = h   + (size_t)s * N_CLASS;
    float*       od = out + (size_t)d * N_CLASS;
    atomicAdd(&od[lane],      hs[lane]      * norm);
    atomicAdd(&od[lane + 32], hs[lane + 32] * norm);
}

// ---------------- row softmax over 64 classes, one wave per row ----------------

__global__ __launch_bounds__(256)
void gcn_softmax(float* __restrict__ out, int nRows) {
    int w = (int)((blockIdx.x * blockDim.x + threadIdx.x) >> 5);
    if (w >= nRows) return;
    int lane = threadIdx.x & 31;
    float* row = out + (size_t)w * N_CLASS;
    float v0 = row[lane];
    float v1 = row[lane + 32];
    float mx = fmaxf(v0, v1);
#pragma unroll
    for (int off = 16; off > 0; off >>= 1) mx = fmaxf(mx, __shfl_xor(mx, off, 32));
    float e0 = __expf(v0 - mx);
    float e1 = __expf(v1 - mx);
    float s = e0 + e1;
#pragma unroll
    for (int off = 16; off > 0; off >>= 1) s += __shfl_xor(s, off, 32);
    float inv = 1.0f / s;
    row[lane]      = e0 * inv;
    row[lane + 32] = e1 * inv;
}

// ---------------- launcher ----------------

static inline size_t align256(size_t v) { return (v + 255) & ~(size_t)255; }

extern "C" void kernel_launch(void* const* d_in, const int* in_sizes, int n_in,
                              void* d_out, int out_size, void* d_ws, size_t ws_size,
                              hipStream_t stream) {
    const float* x    = (const float*)d_in[0];
    const int*   ei   = (const int*)d_in[1];     // edge_index flat: [0:E]=src, [E:2E]=dst
    const float* W    = (const float*)d_in[2];
    const float* bias = (const float*)d_in[3];
    float*       out  = (float*)d_out;

    const int nNodes = in_sizes[0] / N_HIDDEN;   // 50000
    const int nEdges = in_sizes[1] / 2;          // 1600000

    // workspace layout
    size_t hBytes   = (size_t)nNodes * N_CLASS * sizeof(float);
    size_t degOff   = align256(hBytes);
    size_t wOff     = align256(degOff + (size_t)nNodes * sizeof(float));
    float*  h   = (float*)d_ws;
    float*  deg = (float*)((char*)d_ws + degOff);     // becomes dis (rsqrt in place)
    __bf16* Wcm = (__bf16*)((char*)d_ws + wOff);

    const int T = 256;
    // 1) degrees (incl. self loop) -> deg^{-1/2}
    gcn_deg_init <<<(nNodes + T - 1) / T, T, 0, stream>>>(deg, nNodes);
    gcn_deg_edges<<<(nEdges + T - 1) / T, T, 0, stream>>>(ei, deg, nEdges);
    gcn_deg_rsqrt<<<(nNodes + T - 1) / T, T, 0, stream>>>(deg, nNodes);

    // 2) W -> bf16 column-major
    gcn_wcvt<<<(N_HIDDEN * N_CLASS) / T, T, 0, stream>>>(W, Wcm);

    // 3) h = x @ W  (WMMA bf16, 16x64 tile per wave, 8 waves/block)
    int gemmBlocks = (nNodes + 127) / 128;
    gcn_gemm<<<gemmBlocks, T, 0, stream>>>(x, Wcm, h, nNodes);

    // 4) out = h * dis^2 + bias  (self-loop term)
    int total = nNodes * N_CLASS;
    gcn_out_init<<<(total + T - 1) / T, T, 0, stream>>>(h, deg, bias, out, total);

    // 5) edge scatter-add (wave per edge)
    long long scatterThreads = (long long)nEdges * 32;
    gcn_scatter<<<(int)((scatterThreads + T - 1) / T), T, 0, stream>>>(ei, h, deg, out, nEdges);

    // 6) row softmax (wave per row)
    long long smThreads = (long long)nNodes * 32;
    gcn_softmax<<<(int)((smThreads + T - 1) / T), T, 0, stream>>>(out, nNodes);
}
